// LinearWaveAttention_73186242724052
// MI455X (gfx1250) — matrix operand order repaired and checked by reading, not compiled, not run on hardware
//
#include <hip/hip_runtime.h>
#include <hip/hip_bf16.h>
#include <math.h>

// ---------------------------------------------------------------------------
// LinearWaveAttention for MI455X (gfx1250, wave32, WMMA)
//
//   1) split/convert: x f32 -> real,imag bf16; weights -> bf16
//   2) proj_dual<FMAP>: Q = fmap(real@Wq, imag@Wq), K = fmap(real@Wk, imag@Wk),
//      V = real@Wv  (template removes the runtime branch -> no acc copies)
//   3) parallel chunked scan (chunk C=64):
//        S1: G[bh][c] = Kc^T Vc              (4096 blocks, WMMA)
//        S2: exclusive prefix over c of G    (64 blocks, f32 elementwise)
//        S3: O = mask(Qc Kc^T) Vc + Qc G     (4096 blocks, WMMA)
//   4) out_proj: out = Attn @ Wo + bo (f32)
//
// Fragment loads are vectorized: A-fragments are two contiguous b128 LDS
// reads; B-fragments read n-major (transposed-staged) tiles so each lane's
// 16 K-elements are contiguous -> two b128 reads. No scalar assembly movs.
// ---------------------------------------------------------------------------

#define HN   1024
#define NH   16
#define HD   64
#define BB   4
#define SS   4096
#define BS   (BB * SS)      // 16384 rows
#define NC   (SS / 64)      // 64 chunks per sequence
#define EPSW 1e-6f

typedef __attribute__((ext_vector_type(16))) __bf16 v16bf;
typedef __attribute__((ext_vector_type(8)))  __bf16 bfx8;
typedef __attribute__((ext_vector_type(8)))  float  v8f;

#define CAT16(lo, hi) __builtin_shufflevector((lo), (hi), 0, 1, 2, 3, 4, 5, 6, 7, \
                                              8, 9, 10, 11, 12, 13, 14, 15)

// ---------------- fragment loaders (wave32 WMMA VGPR layouts) ---------------
// A (16x32 bf16) from row-major tile T[m][k], row stride ld:
// lane&15 = M, h = lane>>4; elements e <-> k = ko + h*16 + e (contiguous 16)
// => two contiguous 16B loads.
__device__ inline v16bf ldsA(const __bf16* T, int ld, int mb, int ko, int lane) {
    int m = mb + (lane & 15), h = lane >> 4;
    bfx8 lo = *(const bfx8*)&T[m * ld + ko + h * 16];
    bfx8 hi = *(const bfx8*)&T[m * ld + ko + h * 16 + 8];
    return CAT16(lo, hi);
}
// B (32x16 bf16) from an n-major tile Tt[n][k], row stride ld:
// lane&15 = N, h = lane>>4; elements e <-> k = ko + h*16 + e (contiguous 16).
__device__ inline v16bf ldsBn(const __bf16* Tt, int ld, int nb, int ko, int lane) {
    int n = nb + (lane & 15), h = lane >> 4;
    bfx8 lo = *(const bfx8*)&Tt[n * ld + ko + h * 16];
    bfx8 hi = *(const bfx8*)&Tt[n * ld + ko + h * 16 + 8];
    return CAT16(lo, hi);
}

#define WMMA_BF16(A, B, C) \
    __builtin_amdgcn_wmma_f32_16x16x32_bf16(false, (A), false, (B), (short)0, (C), false, false)

__device__ inline float wave_fmap(float re, float im) {
    float amp = sqrtf(re * re + im * im + EPSW);
    float ph  = atan2f(im, re);
    float elu = amp > 0.f ? amp : expm1f(amp);
    return (elu + 1.f) * cosf(sinf(ph));
}

// -------------------------- conversion kernels ------------------------------
__global__ void split_convert_x(const float* __restrict__ x,
                                __bf16* __restrict__ re, __bf16* __restrict__ im) {
    int idx = blockIdx.x * blockDim.x + threadIdx.x;
    if (idx >= BS * HN) return;
    int r = idx / HN, c = idx % HN;
    re[idx] = (__bf16)x[(size_t)r * 2 * HN + c];
    im[idx] = (__bf16)x[(size_t)r * 2 * HN + HN + c];
}

__global__ void convert_f32_bf16(const float* __restrict__ s, __bf16* __restrict__ d, int n) {
    int idx = blockIdx.x * blockDim.x + threadIdx.x;
    if (idx < n) d[idx] = (__bf16)s[idx];
}

// ------------- fused dual projection + wave feature map ---------------------
// 64x64 tile / 128-thread block (4 waves x 16-row strips); K staged 32/iter.
// W tile staged n-major so B-fragment reads are contiguous.
template <bool FMAP>
__global__ __launch_bounds__(128)
void proj_dual_kernel(const __bf16* __restrict__ Ar, const __bf16* __restrict__ Ai,
                      const __bf16* __restrict__ W, __bf16* __restrict__ out) {
    __shared__ __attribute__((aligned(16))) __bf16 Asr[64 * 32];
    __shared__ __attribute__((aligned(16))) __bf16 Asi[64 * 32];
    __shared__ __attribute__((aligned(16))) __bf16 BsT[64 * 32];   // [n][k]

    const int tid = threadIdx.x, wave = tid >> 5, lane = tid & 31;
    const int m0 = blockIdx.x * 64, n0 = blockIdx.y * 64;
    const int mb = wave * 16;

    v8f accR[4] = {}, accI[4] = {};

    for (int kk = 0; kk < HN; kk += 32) {
        __syncthreads();
#pragma unroll
        for (int i = 0; i < 2; ++i) {
            int e  = (tid + i * 128) * 8;          // 2048 bf16 per tile
            int rA = e >> 5, cA = e & 31;
            *(bfx8*)&Asr[e] = *(const bfx8*)&Ar[(size_t)(m0 + rA) * HN + kk + cA];
            if constexpr (FMAP)
                *(bfx8*)&Asi[e] = *(const bfx8*)&Ai[(size_t)(m0 + rA) * HN + kk + cA];
            int rB = e >> 6, cB = e & 63;          // W global row kk+rB, col n0+cB
            bfx8 w = *(const bfx8*)&W[(size_t)(kk + rB) * HN + n0 + cB];
#pragma unroll
            for (int u = 0; u < 8; ++u) BsT[(cB + u) * 32 + rB] = w[u];
        }
        __syncthreads();

        v16bf ar = ldsA(Asr, 32, mb, 0, lane);
        v16bf ai;
        if constexpr (FMAP) ai = ldsA(Asi, 32, mb, 0, lane);
#pragma unroll
        for (int j = 0; j < 4; ++j) {
            v16bf b = ldsBn(BsT, 32, 16 * j, 0, lane);
            accR[j] = WMMA_BF16(ar, b, accR[j]);
            if constexpr (FMAP) accI[j] = WMMA_BF16(ai, b, accI[j]);
        }
    }

    const int h = lane >> 4, nl = lane & 15;
#pragma unroll
    for (int j = 0; j < 4; ++j)
#pragma unroll
        for (int r = 0; r < 8; ++r) {
            int m = m0 + mb + r + 8 * h;
            int n = n0 + 16 * j + nl;
            float o;
            if constexpr (FMAP) o = wave_fmap(accR[j][r], accI[j][r]);
            else                o = accR[j][r];
            out[(size_t)m * HN + n] = (__bf16)o;
        }
}

// ---------------- S1: per-chunk G = Kc^T Vc  (fully parallel) ---------------
// block = (bh, c); K staged d-major (transposed) for the A^T operand,
// V staged e-major for the B operand.
__global__ __launch_bounds__(128)
void chunk_kv_kernel(const __bf16* __restrict__ Km, const __bf16* __restrict__ V,
                     float* __restrict__ G) {
    __shared__ __attribute__((aligned(16))) __bf16 KsT[64 * 64];  // [d][s]
    __shared__ __attribute__((aligned(16))) __bf16 VsT[64 * 64];  // [e][s]

    const int tid = threadIdx.x, wave = tid >> 5, lane = tid & 31;
    const int bc = blockIdx.x, bh = bc >> 6, c = bc & 63;
    const int b = bh >> 4, hh = bh & 15;
    const size_t gr = (size_t)b * SS + (size_t)c * 64;
    const int col0 = hh * HD;

    const int mrow = tid >> 1, cch = (tid & 1) * 32;     // row s, col-half
    const size_t go = (gr + mrow) * HN + col0 + cch;
#pragma unroll
    for (int i = 0; i < 4; ++i) {
        bfx8 kv = *(const bfx8*)&Km[go + i * 8];
        bfx8 vv = *(const bfx8*)&V[go + i * 8];
#pragma unroll
        for (int u = 0; u < 8; ++u) {
            KsT[(cch + i * 8 + u) * 64 + mrow] = kv[u];
            VsT[(cch + i * 8 + u) * 64 + mrow] = vv[u];
        }
    }
    __syncthreads();

    const int mb = wave * 16;                             // d-strip
    v16bf a0 = ldsA(KsT, 64, mb, 0, lane);                // A[d][s]
    v16bf a1 = ldsA(KsT, 64, mb, 32, lane);
    float* Gp = G + (size_t)bc * 4096;
    const int h = lane >> 4, nl = lane & 15;
#pragma unroll
    for (int j = 0; j < 4; ++j) {
        v8f acc = {};
        acc = WMMA_BF16(a0, ldsBn(VsT, 64, 16 * j, 0,  lane), acc);
        acc = WMMA_BF16(a1, ldsBn(VsT, 64, 16 * j, 32, lane), acc);
#pragma unroll
        for (int r = 0; r < 8; ++r)
            Gp[(mb + r + 8 * h) * 64 + 16 * j + nl] = acc[r];
    }
}

// ---------------- S2: exclusive prefix over chunks (tiny serial part) -------
__global__ void prefix_state_kernel(float* __restrict__ G) {
    const int bh = blockIdx.x;
    for (int e = threadIdx.x; e < 4096; e += blockDim.x) {
        float* p = G + (size_t)bh * NC * 4096 + e;
        float acc = 0.f;
        for (int c = 0; c < NC; ++c) {
            float t = p[(size_t)c * 4096];
            p[(size_t)c * 4096] = acc;
            acc += t;
        }
    }
}

// ---------------- S3: O = mask(Qc Kc^T) Vc + Qc State  (parallel) -----------
__global__ __launch_bounds__(128)
void chunk_out_kernel(const __bf16* __restrict__ Q, const __bf16* __restrict__ Km,
                      const __bf16* __restrict__ V, const float* __restrict__ G,
                      __bf16* __restrict__ O) {
    __shared__ __attribute__((aligned(16))) __bf16 Qs [64 * 64];  // [t][d]
    __shared__ __attribute__((aligned(16))) __bf16 Ks [64 * 64];  // [s][d]
    __shared__ __attribute__((aligned(16))) __bf16 VsT[64 * 64];  // [e][s]
    __shared__ __attribute__((aligned(16))) __bf16 SbT[64 * 64];  // [e][d]
    __shared__ __attribute__((aligned(16))) __bf16 Ps [64 * 64];  // [t][s]

    const int tid = threadIdx.x, wave = tid >> 5, lane = tid & 31;
    const int bc = blockIdx.x, bh = bc >> 6, c = bc & 63;
    const int b = bh >> 4, hh = bh & 15;
    const size_t gr = (size_t)b * SS + (size_t)c * 64;
    const int col0 = hh * HD;

    const int mrow = tid >> 1, cch = (tid & 1) * 32;
    const size_t go = (gr + mrow) * HN + col0 + cch;
#pragma unroll
    for (int i = 0; i < 4; ++i) {
        *(bfx8*)&Qs[mrow * 64 + cch + i * 8] = *(const bfx8*)&Q [go + i * 8];
        *(bfx8*)&Ks[mrow * 64 + cch + i * 8] = *(const bfx8*)&Km[go + i * 8];
        bfx8 vv = *(const bfx8*)&V[go + i * 8];
#pragma unroll
        for (int u = 0; u < 8; ++u) VsT[(cch + i * 8 + u) * 64 + mrow] = vv[u];
    }
    // exclusive-prefix state for this chunk, staged e-major as bf16
    for (int i = tid; i < 4096; i += 128) {
        int d = i >> 6, e = i & 63;
        SbT[e * 64 + d] = (__bf16)G[(size_t)bc * 4096 + i];
    }
    __syncthreads();

    const int mb = wave * 16;
    const int h = lane >> 4, nl = lane & 15;
    v16bf q0 = ldsA(Qs, 64, mb, 0, lane);
    v16bf q1 = ldsA(Qs, 64, mb, 32, lane);

    // P = mask(Qc Kc^T): B[k=d][n=s] = Ks[s][d] -> n-major read of Ks
#pragma unroll
    for (int j = 0; j < 4; ++j) {
        v8f acc = {};
        acc = WMMA_BF16(q0, ldsBn(Ks, 64, 16 * j, 0,  lane), acc);
        acc = WMMA_BF16(q1, ldsBn(Ks, 64, 16 * j, 32, lane), acc);
#pragma unroll
        for (int r = 0; r < 8; ++r) {
            int m = mb + r + 8 * h, n = 16 * j + nl;
            float pv = (n <= m) ? acc[r] : 0.f;       // causal incl. diagonal
            Ps[m * 64 + n] = (__bf16)pv;
        }
    }
    __syncthreads();

    // O = P Vc + Qc State
    v16bf p0 = ldsA(Ps, 64, mb, 0, lane);
    v16bf p1 = ldsA(Ps, 64, mb, 32, lane);
#pragma unroll
    for (int j = 0; j < 4; ++j) {
        v8f acc = {};
        acc = WMMA_BF16(p0, ldsBn(VsT, 64, 16 * j, 0,  lane), acc);
        acc = WMMA_BF16(p1, ldsBn(VsT, 64, 16 * j, 32, lane), acc);
        acc = WMMA_BF16(q0, ldsBn(SbT, 64, 16 * j, 0,  lane), acc);
        acc = WMMA_BF16(q1, ldsBn(SbT, 64, 16 * j, 32, lane), acc);
#pragma unroll
        for (int r = 0; r < 8; ++r) {
            size_t row = gr + mb + r + 8 * h;
            O[row * HN + col0 + 16 * j + nl] = (__bf16)acc[r];
        }
    }
}

// -------------------- output projection: Attn @ Wo + bo ---------------------
__global__ __launch_bounds__(128)
void out_proj_kernel(const __bf16* __restrict__ A, const __bf16* __restrict__ W,
                     const float* __restrict__ bias, float* __restrict__ out) {
    __shared__ __attribute__((aligned(16))) __bf16 As [64 * 32];
    __shared__ __attribute__((aligned(16))) __bf16 BsT[64 * 32];  // [n][k]
    const int tid = threadIdx.x, wave = tid >> 5, lane = tid & 31;
    const int m0 = blockIdx.x * 64, n0 = blockIdx.y * 64;
    const int mb = wave * 16;

    v8f acc[4] = {};
    for (int kk = 0; kk < HN; kk += 32) {
        __syncthreads();
#pragma unroll
        for (int i = 0; i < 2; ++i) {
            int e  = (tid + i * 128) * 8;
            int rA = e >> 5, cA = e & 31;
            *(bfx8*)&As[e] = *(const bfx8*)&A[(size_t)(m0 + rA) * HN + kk + cA];
            int rB = e >> 6, cB = e & 63;
            bfx8 w = *(const bfx8*)&W[(size_t)(kk + rB) * HN + n0 + cB];
#pragma unroll
            for (int u = 0; u < 8; ++u) BsT[(cB + u) * 32 + rB] = w[u];
        }
        __syncthreads();
        v16bf a = ldsA(As, 32, mb, 0, lane);
#pragma unroll
        for (int j = 0; j < 4; ++j)
            acc[j] = WMMA_BF16(a, ldsBn(BsT, 32, 16 * j, 0, lane), acc[j]);
    }

    const int h = lane >> 4, nl = lane & 15;
#pragma unroll
    for (int j = 0; j < 4; ++j)
#pragma unroll
        for (int r = 0; r < 8; ++r) {
            int m = m0 + mb + r + 8 * h;
            int n = n0 + 16 * j + nl;
            out[(size_t)m * HN + n] = acc[j][r] + bias[n];
        }
}

// ------------------------------- launcher -----------------------------------
extern "C" void kernel_launch(void* const* d_in, const int* in_sizes, int n_in,
                              void* d_out, int out_size, void* d_ws, size_t ws_size,
                              hipStream_t stream) {
    const float* x  = (const float*)d_in[0];
    const float* Wq = (const float*)d_in[1];
    const float* Wk = (const float*)d_in[2];
    const float* Wv = (const float*)d_in[3];
    const float* Wo = (const float*)d_in[4];
    const float* bo = (const float*)d_in[5];

    char* ws = (char*)d_ws;
    const size_t SZ_ACT = (size_t)BS * HN * sizeof(__bf16);           // 32 MB
    const size_t SZ_W   = (size_t)HN * HN * sizeof(__bf16);           //  2 MB
    const size_t SZ_G   = (size_t)BB * NH * NC * HD * HD * sizeof(float); // 64 MB
    size_t off = 0;
    __bf16* rb  = (__bf16*)(ws + off); off += SZ_ACT;
    __bf16* ib  = (__bf16*)(ws + off); off += SZ_ACT;
    __bf16* wqb = (__bf16*)(ws + off); off += SZ_W;
    __bf16* wkb = (__bf16*)(ws + off); off += SZ_W;
    __bf16* wvb = (__bf16*)(ws + off); off += SZ_W;
    __bf16* wob = (__bf16*)(ws + off); off += SZ_W;
    __bf16* Qb  = (__bf16*)(ws + off); off += SZ_ACT;
    __bf16* Kb  = (__bf16*)(ws + off); off += SZ_ACT;
    __bf16* Vb  = (__bf16*)(ws + off); off += SZ_ACT;
    __bf16* Ab  = (__bf16*)(ws + off); off += SZ_ACT;
    float*  Gb  = (float*) (ws + off); off += SZ_G;

    split_convert_x<<<(BS * HN + 255) / 256, 256, 0, stream>>>(x, rb, ib);
    convert_f32_bf16<<<(HN * HN + 255) / 256, 256, 0, stream>>>(Wq, wqb, HN * HN);
    convert_f32_bf16<<<(HN * HN + 255) / 256, 256, 0, stream>>>(Wk, wkb, HN * HN);
    convert_f32_bf16<<<(HN * HN + 255) / 256, 256, 0, stream>>>(Wv, wvb, HN * HN);
    convert_f32_bf16<<<(HN * HN + 255) / 256, 256, 0, stream>>>(Wo, wob, HN * HN);

    dim3 g(BS / 64, HN / 64);
    proj_dual_kernel<true ><<<g, 128, 0, stream>>>(rb, ib, wqb, Qb);
    proj_dual_kernel<true ><<<g, 128, 0, stream>>>(rb, ib, wkb, Kb);
    proj_dual_kernel<false><<<g, 128, 0, stream>>>(rb, ib, wvb, Vb);

    chunk_kv_kernel   <<<BB * NH * NC, 128, 0, stream>>>(Kb, Vb, Gb);
    prefix_state_kernel<<<BB * NH, 256, 0, stream>>>(Gb);
    chunk_out_kernel  <<<BB * NH * NC, 128, 0, stream>>>(Qb, Kb, Vb, Gb, Ab);

    out_proj_kernel<<<g, 128, 0, stream>>>(Ab, wob, bo, (float*)d_out);
}
